// ALiBiAttention_20478404067603
// MI455X (gfx1250) — matrix operand order repaired and checked by reading, not compiled, not run on hardware
//
#include <hip/hip_runtime.h>
#include <hip/hip_bf16.h>

#define D_MODEL 1024
#define NHEAD   16
#define HDIM    64
#define SEQ     2048
#define BATCH   2
#define MROWS   (BATCH * SEQ)   // 4096

typedef __attribute__((ext_vector_type(16))) _Float16 v16h;
typedef __attribute__((ext_vector_type(8)))  float    v8f;

union Frag16 { v16h v; _Float16 h[16]; unsigned u[8]; };
union U4     { uint4 q; _Float16 h[8];  unsigned u[4]; };

// -- gfx1250 async global->LDS copy (ASYNCcnt-tracked, no VGPR round trip) --
static __device__ __forceinline__ void async_ld_b128(unsigned lds_addr,
                                                     unsigned long long gaddr)
{
    asm volatile("global_load_async_to_lds_b128 %0, %1, off"
                 :: "v"(lds_addr), "v"(gaddr) : "memory");
}
static __device__ __forceinline__ void wait_async0()
{
    asm volatile("s_wait_asynccnt 0" ::: "memory");
}
static __device__ __forceinline__ void wait_ds0()
{
    asm volatile("s_wait_dscnt 0" ::: "memory");
}
static __device__ __forceinline__ unsigned lds_off(const void* p)
{
    // low 32 bits of a flat pointer into LDS == DS byte offset
    return (unsigned)(unsigned long long)p;
}

// ---------------------------------------------------------------------------
// f32 -> f16 conversion (one-time, memory-bound, negligible vs. WMMA work)
// ---------------------------------------------------------------------------
__global__ void cvt_f32_f16(const float* __restrict__ in, _Float16* __restrict__ out, int n)
{
    int i = blockIdx.x * blockDim.x + threadIdx.x;
    if (i < n) out[i] = (_Float16)in[i];
}

// ---------------------------------------------------------------------------
// Tiled WMMA GEMM (all dims compile-time):
//   C[4096,1024] = A[4096,1024] @ B[1024,1024] + bias
// Block: 256 threads = 8 waves, tile 128x128, K-step 32, register-prefetch
// double buffering (fetch tile k+1 into VGPRs while WMMAs consume tile k).
// MODE 0: scatter f16 output into head-split [B*H, S, 64] layout (Q/K/V)
// MODE 1: write f32 output [M,N] (final projection into d_out)
// ---------------------------------------------------------------------------
template <int MODE>
__global__ __launch_bounds__(256)
void gemm_f16_wmma(const _Float16* __restrict__ A, const _Float16* __restrict__ Bm,
                   const float* __restrict__ bias, void* __restrict__ outp)
{
    constexpr int K = D_MODEL;     // 1024
    constexpr int N = D_MODEL;     // 1024

    __shared__ _Float16 As[128 * 32];   // [row][k]              8 KB
    __shared__ _Float16 Bs[128 * 32];   // [col][k] (transposed) 8 KB

    const int tid  = threadIdx.x;
    const int lane = tid & 31;
    const int wid  = tid >> 5;
    const int wm   = wid & 3;       // 0..3 -> row group of 32
    const int wn   = wid >> 2;      // 0..1 -> col group of 64
    const int rowBase = blockIdx.y * 128;
    const int colBase = blockIdx.x * 128;
    const int l16 = lane & 15;
    const int kg  = lane >> 4;

    // per-thread staging coordinates
    const int ar = tid >> 2, ac = (tid & 3) * 8;     // A: 128 rows x 32 k
    const int bk = tid >> 4, bn = (tid & 15) * 8;    // B: 32 k x 128 n
    const _Float16* Ap0 = A  + (size_t)(rowBase + ar) * K + ac;
    const _Float16* Ap1 = Ap0 + (size_t)64 * K;
    const _Float16* Bp0 = Bm + (size_t)bk * N + colBase + bn;
    const _Float16* Bp1 = Bp0 + (size_t)16 * N;

    v8f acc[2][4];
    #pragma unroll
    for (int i = 0; i < 2; ++i)
        #pragma unroll
        for (int j = 0; j < 4; ++j) acc[i][j] = (v8f)0.0f;

    U4 ra0, ra1, rb0, rb1;
    // prologue fetch: tile 0
    ra0.q = *(const uint4*)(Ap0);
    ra1.q = *(const uint4*)(Ap1);
    rb0.q = *(const uint4*)(Bp0);
    rb1.q = *(const uint4*)(Bp1);

    for (int kt = 0; kt < K; kt += 32) {
        // ---- publish prefetched tile to LDS ----
        *(uint4*)&As[ar * 32 + ac]        = ra0.q;
        *(uint4*)&As[(ar + 64) * 32 + ac] = ra1.q;
        #pragma unroll
        for (int j = 0; j < 8; ++j) {
            Bs[(bn + j) * 32 + bk]      = rb0.h[j];
            Bs[(bn + j) * 32 + bk + 16] = rb1.h[j];
        }
        __syncthreads();

        // ---- prefetch next tile into registers (overlaps WMMA below) ----
        if (kt + 32 < K) {
            ra0.q = *(const uint4*)(Ap0 + kt + 32);
            ra1.q = *(const uint4*)(Ap1 + kt + 32);
            rb0.q = *(const uint4*)(Bp0 + (size_t)(kt + 32) * N);
            rb1.q = *(const uint4*)(Bp1 + (size_t)(kt + 32) * N);
        }

        // ---- fragment gathers A(16x32)/B(32x16), contiguous k = 16*kg + s in both ----
        Frag16 af[2], bf[4];
        #pragma unroll
        for (int fm = 0; fm < 2; ++fm) {
            int row = wm * 32 + fm * 16 + l16;
            #pragma unroll
            for (int g = 0; g < 8; ++g) {
                int kk = 2 * g + 16 * kg;
                af[fm].u[g] = *(const unsigned*)&As[row * 32 + kk];
            }
        }
        #pragma unroll
        for (int fn = 0; fn < 4; ++fn) {
            int col = wn * 64 + fn * 16 + l16;
            #pragma unroll
            for (int g = 0; g < 8; ++g) {
                int kk = 2 * g + 16 * kg;
                bf[fn].u[g] = *(const unsigned*)&Bs[col * 32 + kk];
            }
        }
        #pragma unroll
        for (int fm = 0; fm < 2; ++fm)
            #pragma unroll
            for (int fn = 0; fn < 4; ++fn)
                acc[fm][fn] = __builtin_amdgcn_wmma_f32_16x16x32_f16(
                    false, af[fm].v, false, bf[fn].v, (short)0, acc[fm][fn], false, false);
        __syncthreads();
    }

    // ---- epilogue: bias + store (C layout: lane = col, 8 rows per lane) ----
    #pragma unroll
    for (int fm = 0; fm < 2; ++fm) {
        #pragma unroll
        for (int fn = 0; fn < 4; ++fn) {
            int n = colBase + wn * 64 + fn * 16 + l16;
            float bnv = bias[n];
            #pragma unroll
            for (int r = 0; r < 8; ++r) {
                int m = rowBase + wm * 32 + fm * 16 + 8 * kg + r;
                float val = acc[fm][fn][r] + bnv;
                if (MODE == 0) {
                    int b = m >> 11, s = m & (SEQ - 1);
                    int h = n >> 6,  d = n & (HDIM - 1);
                    ((_Float16*)outp)[(((size_t)(b * NHEAD + h) * SEQ) + s) * HDIM + d] =
                        (_Float16)val;
                } else {
                    ((float*)outp)[(size_t)m * N + n] = val;
                }
            }
        }
    }
}

// ---------------------------------------------------------------------------
// Streaming (flash-style) ALiBi attention.
// Block: 128 threads = 4 waves. grid = (S/64 query tiles, B*H heads).
// Each wave owns 16 query rows. K/V chunks staged with double-buffered
// global_load_async_to_lds_b128 (ASYNCcnt), one barrier per key chunk.
// ---------------------------------------------------------------------------
__global__ __launch_bounds__(128)
void alibi_attn_f16(const _Float16* __restrict__ Q, const _Float16* __restrict__ Km,
                    const _Float16* __restrict__ V, _Float16* __restrict__ O)
{
    __shared__ _Float16 Kc[2 * 64 * 64];   // 16 KB (double-buffered)
    __shared__ _Float16 Vc[2 * 64 * 64];   // 16 KB (double-buffered)
    __shared__ _Float16 Pl[4 * 16 * 64];   //  8 KB, per-wave P relayout buffers

    const int tid  = threadIdx.x;
    const int lane = tid & 31;
    const int wid  = tid >> 5;
    const int l16  = lane & 15;
    const int kg   = lane >> 4;
    const int bh   = blockIdx.y;
    const int h    = bh & (NHEAD - 1);
    const int qbase = blockIdx.x * 64 + wid * 16;   // first query row of this wave

    const float slope = exp2f(-0.5f * (float)(h + 1));   // 2^{-8(h+1)/H}, H=16
    const size_t headOff = (size_t)bh * SEQ * HDIM;
    const _Float16* Kg = Km + headOff;
    const _Float16* Vg = V  + headOff;

    // staging coordinates: 64x64 halves = 512 x 16B, 128 threads x 4 vectors
    const int sr = tid >> 1;              // 0..63   (row within chunk), it adds +? no:
    // idx = tid + it*128 ; r = idx>>3 ; c = (idx&7)*8  -> r = (tid>>3) + it*16
    const int r0 = tid >> 3;
    const int c0 = (tid & 7) * 8;
    (void)sr;

    // issue one 64-key chunk (K and V) into buffer `buf` via async copies
    auto issue_chunk = [&](int kb, int buf) {
        const _Float16* kgp = Kg + (size_t)kb * HDIM;
        const _Float16* vgp = Vg + (size_t)kb * HDIM;
        _Float16* kl = Kc + buf * (64 * 64);
        _Float16* vl = Vc + buf * (64 * 64);
        #pragma unroll
        for (int it = 0; it < 4; ++it) {
            int r = r0 + it * 16;
            async_ld_b128(lds_off(&kl[r * 64 + c0]),
                          (unsigned long long)(kgp + (size_t)r * HDIM + c0));
            async_ld_b128(lds_off(&vl[r * 64 + c0]),
                          (unsigned long long)(vgp + (size_t)r * HDIM + c0));
        }
    };

    // Q fragments: 16 rows x 64 Dh -> two 16x32 A-fragments, loaded once
    Frag16 aq[2];
    #pragma unroll
    for (int ks = 0; ks < 2; ++ks) {
        int row = qbase + l16;
        #pragma unroll
        for (int g = 0; g < 8; ++g) {
            int kk = 2 * g + 16 * kg;
            aq[ks].u[g] = *(const unsigned*)&Q[headOff + (size_t)row * HDIM + ks * 32 + kk];
        }
    }

    v8f acc[4];
    #pragma unroll
    for (int i = 0; i < 4; ++i) acc[i] = (v8f)0.0f;
    float rowM[8], rowS[8];
    #pragma unroll
    for (int r = 0; r < 8; ++r) { rowM[r] = -1e30f; rowS[r] = 0.0f; }

    issue_chunk(0, 0);   // prologue

    for (int kb = 0; kb < SEQ; kb += 64) {
        const int buf = (kb >> 6) & 1;
        wait_async0();        // this wave's chunk landed in LDS
        __syncthreads();      // all waves' copies landed; prev buffer now free
        if (kb + 64 < SEQ) issue_chunk(kb + 64, buf ^ 1);   // overlaps compute

        const _Float16* Kb = Kc + buf * (64 * 64);
        const _Float16* Vb = Vc + buf * (64 * 64);

        // ---- scores: 16 x 64 = Q_tile @ K_chunk^T (contraction over Dh) ----
        v8f sc[4];
        #pragma unroll
        for (int nt = 0; nt < 4; ++nt) {
            sc[nt] = (v8f)0.0f;
            #pragma unroll
            for (int ks = 0; ks < 2; ++ks) {
                Frag16 bkf;
                int key = nt * 16 + l16;
                #pragma unroll
                for (int g = 0; g < 8; ++g) {
                    int d2 = ks * 32 + 2 * g + 16 * kg;
                    bkf.u[g] = *(const unsigned*)&Kb[key * 64 + d2];
                }
                sc[nt] = __builtin_amdgcn_wmma_f32_16x16x32_f16(
                    false, aq[ks].v, false, bkf.v, (short)0, sc[nt], false, false);
            }
        }

        // ---- scale + ALiBi bias, chunk row max ----
        float nm[8];
        #pragma unroll
        for (int r = 0; r < 8; ++r) nm[r] = -1e30f;
        #pragma unroll
        for (int nt = 0; nt < 4; ++nt) {
            #pragma unroll
            for (int r = 0; r < 8; ++r) {
                float i_abs = (float)(qbase + 8 * kg + r);
                float j_abs = (float)(kb + nt * 16 + l16);
                float v = sc[nt][r] * 0.125f - slope * fabsf(i_abs - j_abs);
                sc[nt][r] = v;
                nm[r] = fmaxf(nm[r], v);
            }
        }
        #pragma unroll
        for (int r = 0; r < 8; ++r) {
            #pragma unroll
            for (int off = 1; off < 16; off <<= 1)
                nm[r] = fmaxf(nm[r], __shfl_xor(nm[r], off, 32));
        }

        // ---- online softmax update ----
        float alpha[8], psum[8];
        #pragma unroll
        for (int r = 0; r < 8; ++r) {
            float Mn = fmaxf(rowM[r], nm[r]);
            alpha[r] = __expf(rowM[r] - Mn);
            rowM[r] = Mn;
            psum[r] = 0.0f;
        }
        #pragma unroll
        for (int nt = 0; nt < 4; ++nt) {
            #pragma unroll
            for (int r = 0; r < 8; ++r) {
                float p = __expf(sc[nt][r] - rowM[r]);
                psum[r] += p;
                // relayout C-fragment -> LDS row-major P tile (per-wave region)
                Pl[wid * 1024 + (8 * kg + r) * 64 + nt * 16 + l16] = (_Float16)p;
            }
        }
        #pragma unroll
        for (int r = 0; r < 8; ++r) {
            #pragma unroll
            for (int off = 1; off < 16; off <<= 1)
                psum[r] += __shfl_xor(psum[r], off, 32);
            rowS[r] = rowS[r] * alpha[r] + psum[r];
        }
        #pragma unroll
        for (int dt = 0; dt < 4; ++dt)
            #pragma unroll
            for (int r = 0; r < 8; ++r)
                acc[dt][r] *= alpha[r];

        // wave-private P writes must land before re-reading as A-fragments
        wait_ds0();

        // ---- O += P @ V_chunk (contraction over 64 keys = 2 k-steps) ----
        #pragma unroll
        for (int ks = 0; ks < 2; ++ks) {
            Frag16 ap;
            #pragma unroll
            for (int g = 0; g < 8; ++g) {
                int kk = 2 * g + 16 * kg;
                ap.u[g] = *(const unsigned*)&Pl[wid * 1024 + l16 * 64 + ks * 32 + kk];
            }
            #pragma unroll
            for (int dt = 0; dt < 4; ++dt) {
                Frag16 bv;
                int n = dt * 16 + l16;
                #pragma unroll
                for (int g = 0; g < 8; ++g) {
                    int k0 = ks * 32 + 2 * g + 16 * kg;
                    bv.h[2 * g]     = Vb[k0 * 64 + n];
                    bv.h[2 * g + 1] = Vb[(k0 + 1) * 64 + n];
                }
                acc[dt] = __builtin_amdgcn_wmma_f32_16x16x32_f16(
                    false, ap.v, false, bv.v, (short)0, acc[dt], false, false);
            }
        }
    }

    // ---- normalize and write merged-head layout [B, S, H*Dh] ----
    const int b = bh >> 4;
    #pragma unroll
    for (int dt = 0; dt < 4; ++dt) {
        int d = dt * 16 + l16;
        #pragma unroll
        for (int r = 0; r < 8; ++r) {
            int s = qbase + 8 * kg + r;
            float v = acc[dt][r] / rowS[r];
            O[((size_t)(b * SEQ + s)) * D_MODEL + h * HDIM + d] = (_Float16)v;
        }
    }
}

// ---------------------------------------------------------------------------
// Host-side orchestration (all launches on `stream`, graph-capture safe)
// ---------------------------------------------------------------------------
extern "C" void kernel_launch(void* const* d_in, const int* in_sizes, int n_in,
                              void* d_out, int out_size, void* d_ws, size_t ws_size,
                              hipStream_t stream)
{
    (void)in_sizes; (void)n_in; (void)out_size; (void)ws_size;

    const float* x  = (const float*)d_in[0];
    const float* Wq = (const float*)d_in[1];
    const float* bq = (const float*)d_in[2];
    const float* Wk = (const float*)d_in[3];
    const float* bk = (const float*)d_in[4];
    const float* Wv = (const float*)d_in[5];
    const float* bv = (const float*)d_in[6];
    const float* Wo = (const float*)d_in[7];
    const float* bo = (const float*)d_in[8];

    char* ws = (char*)d_ws;
    _Float16* xh  = (_Float16*)(ws);                            //  0 -  8 MB
    _Float16* wqh = (_Float16*)(ws + ((size_t)8  << 20));       //  8 - 10 MB
    _Float16* wkh = (_Float16*)(ws + ((size_t)10 << 20));       // 10 - 12 MB
    _Float16* wvh = (_Float16*)(ws + ((size_t)12 << 20));       // 12 - 14 MB
    _Float16* woh = (_Float16*)(ws + ((size_t)14 << 20));       // 14 - 16 MB
    _Float16* Qh  = (_Float16*)(ws + ((size_t)16 << 20));       // 16 - 24 MB
    _Float16* Kh  = (_Float16*)(ws + ((size_t)24 << 20));       // 24 - 32 MB
    _Float16* Vh  = (_Float16*)(ws + ((size_t)32 << 20));       // 32 - 40 MB
    _Float16* Ah  = (_Float16*)(ws + ((size_t)40 << 20));       // 40 - 48 MB

    const int nX = MROWS * D_MODEL;        // 4,194,304
    const int nW = D_MODEL * D_MODEL;      // 1,048,576
    cvt_f32_f16<<<(nX + 255) / 256, 256, 0, stream>>>(x,  xh,  nX);
    cvt_f32_f16<<<(nW + 255) / 256, 256, 0, stream>>>(Wq, wqh, nW);
    cvt_f32_f16<<<(nW + 255) / 256, 256, 0, stream>>>(Wk, wkh, nW);
    cvt_f32_f16<<<(nW + 255) / 256, 256, 0, stream>>>(Wv, wvh, nW);
    cvt_f32_f16<<<(nW + 255) / 256, 256, 0, stream>>>(Wo, woh, nW);

    dim3 ggrid(D_MODEL / 128, MROWS / 128);   // (8, 32)
    gemm_f16_wmma<0><<<ggrid, 256, 0, stream>>>(xh, wqh, bq, (void*)Qh);
    gemm_f16_wmma<0><<<ggrid, 256, 0, stream>>>(xh, wkh, bk, (void*)Kh);
    gemm_f16_wmma<0><<<ggrid, 256, 0, stream>>>(xh, wvh, bv, (void*)Vh);

    alibi_attn_f16<<<dim3(SEQ / 64, BATCH * NHEAD), 128, 0, stream>>>(Qh, Kh, Vh, Ah);

    gemm_f16_wmma<1><<<ggrid, 256, 0, stream>>>(Ah, woh, bo, d_out);
}
